// AttentionLayer_37546604102049
// MI455X (gfx1250) — matrix operand order repaired and checked by reading, not compiled, not run on hardware
//
#include <hip/hip_runtime.h>

// ---------------------------------------------------------------------------
// MI455X / gfx1250 attention layer.
//   T=1024, B=8, D=1024, H=16, NH=64.  All matrix math on
//   v_wmma_f32_16x16x32_bf16 (wave32), f32 accumulate, bf16 staging in LDS.
// ---------------------------------------------------------------------------

typedef __attribute__((ext_vector_type(16))) __bf16 v16bf;
typedef __attribute__((ext_vector_type(8)))  __bf16 v8bf;
typedef __attribute__((ext_vector_type(8)))  float  v8f;

#define TT  1024
#define BB  8
#define DD  1024
#define HH  16
#define NHD 64

union Pack4 { __bf16 h[4]; unsigned long long u; };
union Pack8 { __bf16 h[8]; v8bf v; };

__device__ __forceinline__ v8f zero8() {
  v8f z = {0.f, 0.f, 0.f, 0.f, 0.f, 0.f, 0.f, 0.f};
  return z;
}

__device__ __forceinline__ v16bf cat16(v8bf lo, v8bf hi) {
  return __builtin_shufflevector(lo, hi, 0, 1, 2, 3, 4, 5, 6, 7,
                                          8, 9, 10, 11, 12, 13, 14, 15);
}

// A-fragment (16xK=32 bf16, row-major tile in LDS).
// ISA layout: lanes 0-15 -> M=lane, K = 0..7 then 16..23;
//             lanes 16-31 -> M=lane-16, K = 8..15 then 24..31.
__device__ __forceinline__ v16bf frag_a(const __bf16* tile, int stride) {
  const int l = threadIdx.x & 31;
  const int m = l & 15;
  const int h = l >> 4;
  const __bf16* p = tile + m * stride + h * 8;
  const v8bf lo = *reinterpret_cast<const v8bf*>(p);
  const v8bf hi = *reinterpret_cast<const v8bf*>(p + 16);
  return cat16(lo, hi);
}

// B-fragment (K=32 x 16 bf16) read from a TRANSPOSED tile Bt[n][k].
// ISA layout: lane n=lane&15 holds K = (lane>>4)*8 + 0..7 then 16 + (lane>>4)*8 + 0..7.
__device__ __forceinline__ v16bf frag_b(const __bf16* tileT, int stride) {
  const int l = threadIdx.x & 31;
  const int n = l & 15;
  const int h = l >> 4;
  const __bf16* p = tileT + n * stride + h * 8;
  const v8bf lo = *reinterpret_cast<const v8bf*>(p);
  const v8bf hi = *reinterpret_cast<const v8bf*>(p + 16);
  return cat16(lo, hi);
}

__device__ __forceinline__ v8f wmma_bf16(v16bf a, v16bf b, v8f c) {
  return __builtin_amdgcn_wmma_f32_16x16x32_bf16(
      /*neg_a=*/false, a, /*neg_b=*/false, b,
      /*c_mod=*/(short)0, c, /*reuse_a=*/false, /*reuse_b=*/false);
}

// ---------------------------------------------------------------------------
// Kernel 1: per-head projection GEMM.
//   C[r, c] = sum_d X[r, d] * W[c>>6, d, c&63] + bias[c]
//   r = t*B + b (8192 rows), c = h*64 + k (1024 cols).
//   Output scattered to bf16 layout [(b*H + h)][T][64] for the attn kernel.
// ---------------------------------------------------------------------------
__global__ __launch_bounds__(256) void proj_gemm_kernel(
    const float* __restrict__ X, const float* __restrict__ W,
    const float* __restrict__ bias, __bf16* __restrict__ Out) {
  __shared__ __bf16 As[128 * 40];  // A tile 128x32 bf16 (stride 40)
  __shared__ __bf16 Bs[64 * 40];   // Bt tile 64(N) x 32(K) bf16

  const int tid  = threadIdx.x;
  const int wave = tid >> 5;
  const int lane = tid & 31;
  const int mblk = blockIdx.x * 128;
  const int nblk = blockIdx.y * 64;
  const int mw   = (wave & 3) * 32;
  const int nw   = (wave >> 2) * 32;
  const int headW = nblk >> 6;  // one head per 64-wide N block

  v8f acc[2][2];
#pragma unroll
  for (int a = 0; a < 2; ++a)
#pragma unroll
    for (int b = 0; b < 2; ++b) acc[a][b] = zero8();

  for (int kb = 0; kb < DD; kb += 32) {
    // ---- stage A: 128x32 f32 -> bf16 ----
    {
      const int row = tid >> 3;        // 0..31
      const int col = (tid & 7) * 4;   // 0..28
#pragma unroll
      for (int p = 0; p < 4; ++p) {
        const int r = row + p * 32;
        const float4 v = *reinterpret_cast<const float4*>(
            &X[(size_t)(mblk + r) * DD + kb + col]);
        Pack4 pk;
        pk.h[0] = (__bf16)v.x; pk.h[1] = (__bf16)v.y;
        pk.h[2] = (__bf16)v.z; pk.h[3] = (__bf16)v.w;
        *reinterpret_cast<unsigned long long*>(&As[r * 40 + col]) = pk.u;
      }
      if (kb + 32 < DD)  // global_prefetch_b8 for next K slab
        __builtin_prefetch(&X[(size_t)(mblk + row) * DD + kb + 32 + col], 0, 1);
    }
    // ---- stage Bt: Bs[n][k] = W[head, kb+k, n] ----
    {
      const int nn  = tid & 63;
      const int kk0 = (tid >> 6) * 8;
      Pack8 pk;
#pragma unroll
      for (int j = 0; j < 8; ++j) {
        const float w = W[((size_t)headW * DD + kb + kk0 + j) * NHD + nn];
        pk.h[j] = (__bf16)w;
      }
      *reinterpret_cast<v8bf*>(&Bs[nn * 40 + kk0]) = pk.v;
    }
    __syncthreads();

    v16bf af[2];
#pragma unroll
    for (int mi = 0; mi < 2; ++mi)
      af[mi] = frag_a(&As[(mw + mi * 16) * 40], 40);
#pragma unroll
    for (int ni = 0; ni < 2; ++ni) {
      const v16bf bf = frag_b(&Bs[(nw + ni * 16) * 40], 40);
#pragma unroll
      for (int mi = 0; mi < 2; ++mi)
        acc[mi][ni] = wmma_bf16(af[mi], bf, acc[mi][ni]);
    }
    __syncthreads();
  }

  // ---- epilogue: bias + scatter to [(b*H+h)][T][NH] bf16 ----
#pragma unroll
  for (int mi = 0; mi < 2; ++mi)
#pragma unroll
    for (int ni = 0; ni < 2; ++ni) {
      const int col = nblk + nw + ni * 16 + (lane & 15);
      const float bcol = bias[col];
      const int hW = col >> 6;
      const int kc = col & 63;
#pragma unroll
      for (int i = 0; i < 8; ++i) {
        const int row = mblk + mw + mi * 16 + (lane >> 4) * 8 + i;
        const int t = row >> 3;  // B == 8
        const int b = row & 7;
        Out[(((size_t)b * HH + hW) * TT + t) * NHD + kc] =
            (__bf16)(acc[mi][ni][i] + bcol);
      }
    }
}

// ---------------------------------------------------------------------------
// Kernel 2: flash-attention per (b,h).  128 queries/block, 4 waves x 32 rows,
// key blocks of 32, online softmax (1 row per thread), causal mask.
// ---------------------------------------------------------------------------
__global__ __launch_bounds__(128) void attn_kernel(
    const __bf16* __restrict__ Qg, const __bf16* __restrict__ Kg,
    const __bf16* __restrict__ Vg, __bf16* __restrict__ Og,
    const int* __restrict__ maskp) {
  __shared__ __bf16 qbuf[128 * 72];  // queries [128][64] (stride 72)
  __shared__ __bf16 kbuf[32 * 72];   // keys [32 keys][64 dims] == Bt for QK^T
  __shared__ __bf16 vtbuf[64 * 40];  // V transposed [64 dims][32 keys]
  __shared__ float  sbuf[128 * 36];  // scores [128][32] f32
  __shared__ __bf16 pbuf[128 * 40];  // probs  [128][32] bf16
  __shared__ float  abuf[128];
  __shared__ float  lbuf[128];

  const int tid   = threadIdx.x;
  const int wave  = tid >> 5;
  const int lane  = tid & 31;
  const int bh    = blockIdx.y;           // b*H + h
  const int qbase = blockIdx.x * 128;
  const size_t off = (size_t)bh * TT * NHD;
  const __bf16* Q = Qg + off;
  const __bf16* K = Kg + off;
  const __bf16* V = Vg + off;
  const int mask = *maskp;

  // ---- load the block's Q tile (bf16, row-major) ----
  {
    const v8bf* src =
        reinterpret_cast<const v8bf*>(&Q[(size_t)(qbase + tid) * NHD]);
#pragma unroll
    for (int j = 0; j < 8; ++j)
      *reinterpret_cast<v8bf*>(&qbuf[tid * 72 + j * 8]) = src[j];
  }
  __syncthreads();

  float m_run = -3.0e38f, l_run = 0.0f;
  v8f ofr[2][4];
#pragma unroll
  for (int mi = 0; mi < 2; ++mi)
#pragma unroll
    for (int di = 0; di < 4; ++di) ofr[mi][di] = zero8();

  const int kend = mask ? (qbase + 128) : TT;
  for (int kb = 0; kb < kend; kb += 32) {
    // ---- cooperative K / V^T staging (32 keys x 64 dims) ----
    {
      const int s  = tid >> 2;
      const int c0 = (tid & 3) * 16;
      const v8bf* ks =
          reinterpret_cast<const v8bf*>(&K[(size_t)(kb + s) * NHD + c0]);
      *reinterpret_cast<v8bf*>(&kbuf[s * 72 + c0])     = ks[0];
      *reinterpret_cast<v8bf*>(&kbuf[s * 72 + c0 + 8]) = ks[1];
      const v8bf* vs =
          reinterpret_cast<const v8bf*>(&V[(size_t)(kb + s) * NHD + c0]);
      const v8bf v0 = vs[0], v1 = vs[1];
#pragma unroll
      for (int j = 0; j < 8; ++j) vtbuf[(c0 + j) * 40 + s] = v0[j];
#pragma unroll
      for (int j = 0; j < 8; ++j) vtbuf[(c0 + 8 + j) * 40 + s] = v1[j];
    }
    __syncthreads();

    const bool wskip = mask && (kb > qbase + wave * 32 + 31);
    if (!wskip) {
      // ---- S = (Q K^T) * scale, causal mask, spill to sbuf ----
      v16bf aq[2][2];
#pragma unroll
      for (int mi = 0; mi < 2; ++mi)
#pragma unroll
        for (int ks2 = 0; ks2 < 2; ++ks2)
          aq[mi][ks2] =
              frag_a(&qbuf[(wave * 32 + mi * 16) * 72 + ks2 * 32], 72);
#pragma unroll
      for (int ni = 0; ni < 2; ++ni) {
        const v16bf bk0 = frag_b(&kbuf[(ni * 16) * 72], 72);
        const v16bf bk1 = frag_b(&kbuf[(ni * 16) * 72 + 32], 72);
#pragma unroll
        for (int mi = 0; mi < 2; ++mi) {
          v8f s8 = zero8();
          s8 = wmma_bf16(aq[mi][0], bk0, s8);
          s8 = wmma_bf16(aq[mi][1], bk1, s8);
          const int r0 = wave * 32 + mi * 16 + (lane >> 4) * 8;
          const int cc = ni * 16 + (lane & 15);
          const int sg = kb + cc;
#pragma unroll
          for (int i = 0; i < 8; ++i) {
            float sv = s8[i] * 0.125f;  // 1/sqrt(64)
            if (mask && sg > qbase + r0 + i) sv = -1.0e9f;
            sbuf[(r0 + i) * 36 + cc] = sv;
          }
        }
      }
    }
    __syncthreads();

    // ---- online softmax: thread `tid` owns query row `tid` ----
    const bool rskip = mask && (kb > qbase + tid);
    if (!rskip) {
      const float4* sp = reinterpret_cast<const float4*>(&sbuf[tid * 36]);
      float rmax = -3.0e38f;
#pragma unroll
      for (int j = 0; j < 8; ++j) {
        const float4 s4 = sp[j];
        rmax = fmaxf(rmax, fmaxf(fmaxf(s4.x, s4.y), fmaxf(s4.z, s4.w)));
      }
      const float mnew  = fmaxf(m_run, rmax);
      const float alpha = __expf(m_run - mnew);
      float psum = 0.0f;
#pragma unroll
      for (int j = 0; j < 8; ++j) {
        const float4 s4 = sp[j];
        const float p0 = __expf(s4.x - mnew);
        const float p1 = __expf(s4.y - mnew);
        const float p2 = __expf(s4.z - mnew);
        const float p3 = __expf(s4.w - mnew);
        psum += p0 + p1 + p2 + p3;
        Pack4 pk;
        pk.h[0] = (__bf16)p0; pk.h[1] = (__bf16)p1;
        pk.h[2] = (__bf16)p2; pk.h[3] = (__bf16)p3;
        *reinterpret_cast<unsigned long long*>(&pbuf[tid * 40 + j * 4]) = pk.u;
      }
      l_run = l_run * alpha + psum;
      m_run = mnew;
      abuf[tid] = alpha;
    } else {
      abuf[tid] = 1.0f;
    }
    __syncthreads();

    // ---- O = O*alpha + P @ V ----
    if (!wskip) {
      float asc[2][8];
#pragma unroll
      for (int mi = 0; mi < 2; ++mi)
#pragma unroll
        for (int i = 0; i < 8; ++i)
          asc[mi][i] = abuf[wave * 32 + mi * 16 + (lane >> 4) * 8 + i];
      v16bf ap[2];
#pragma unroll
      for (int mi = 0; mi < 2; ++mi)
        ap[mi] = frag_a(&pbuf[(wave * 32 + mi * 16) * 40], 40);
#pragma unroll
      for (int di = 0; di < 4; ++di) {
        const v16bf bv = frag_b(&vtbuf[(di * 16) * 40], 40);
#pragma unroll
        for (int mi = 0; mi < 2; ++mi) {
          v8f o = ofr[mi][di];
#pragma unroll
          for (int i = 0; i < 8; ++i) o[i] *= asc[mi][i];
          ofr[mi][di] = wmma_bf16(ap[mi], bv, o);
        }
      }
    }
    __syncthreads();
  }

  lbuf[tid] = 1.0f / l_run;
  __syncthreads();

  // ---- normalize + store to [T*B][D] bf16 (concat heads) ----
  const int b = bh >> 4;   // H == 16
  const int h = bh & 15;
#pragma unroll
  for (int mi = 0; mi < 2; ++mi)
#pragma unroll
    for (int di = 0; di < 4; ++di) {
      const int dim = di * 16 + (lane & 15);
#pragma unroll
      for (int i = 0; i < 8; ++i) {
        const int r = wave * 32 + mi * 16 + (lane >> 4) * 8 + i;
        const float linv = lbuf[r];
        const int t = qbase + r;
        Og[((size_t)t * BB + b) * DD + h * NHD + dim] =
            (__bf16)(ofr[mi][di][i] * linv);
      }
    }
}

// ---------------------------------------------------------------------------
// Kernel 3: output projection.  Out[r, c] = sum_d A[r, d]*Wo[c, d] + bo[c].
// A is bf16 [8192][1024]; Wo row-major IS the transposed-B layout. f32 out.
// ---------------------------------------------------------------------------
__global__ __launch_bounds__(256) void out_gemm_kernel(
    const __bf16* __restrict__ A, const float* __restrict__ Wo,
    const float* __restrict__ bo, float* __restrict__ Out) {
  __shared__ __bf16 As[128 * 40];
  __shared__ __bf16 Bs[64 * 40];

  const int tid  = threadIdx.x;
  const int wave = tid >> 5;
  const int lane = tid & 31;
  const int mblk = blockIdx.x * 128;
  const int nblk = blockIdx.y * 64;
  const int mw   = (wave & 3) * 32;
  const int nw   = (wave >> 2) * 32;

  v8f acc[2][2];
#pragma unroll
  for (int a = 0; a < 2; ++a)
#pragma unroll
    for (int b = 0; b < 2; ++b) acc[a][b] = zero8();

  for (int kb = 0; kb < DD; kb += 32) {
    // ---- stage A (already bf16, direct copy) ----
    {
      const int row = tid >> 1;
      const int c0  = (tid & 1) * 16;
      const v8bf* src = reinterpret_cast<const v8bf*>(
          &A[(size_t)(mblk + row) * DD + kb + c0]);
      *reinterpret_cast<v8bf*>(&As[row * 40 + c0])     = src[0];
      *reinterpret_cast<v8bf*>(&As[row * 40 + c0 + 8]) = src[1];
      if (kb + 32 < DD)
        __builtin_prefetch(&A[(size_t)(mblk + row) * DD + kb + 32 + c0], 0, 1);
    }
    // ---- stage Bt: Bs[n][k] = Wo[nblk+n][kb+k] (contiguous) ----
    {
      const int nn  = tid & 63;
      const int kk0 = (tid >> 6) * 8;
      const float4 w0 = *reinterpret_cast<const float4*>(
          &Wo[(size_t)(nblk + nn) * DD + kb + kk0]);
      const float4 w1 = *reinterpret_cast<const float4*>(
          &Wo[(size_t)(nblk + nn) * DD + kb + kk0 + 4]);
      Pack8 pk;
      pk.h[0] = (__bf16)w0.x; pk.h[1] = (__bf16)w0.y;
      pk.h[2] = (__bf16)w0.z; pk.h[3] = (__bf16)w0.w;
      pk.h[4] = (__bf16)w1.x; pk.h[5] = (__bf16)w1.y;
      pk.h[6] = (__bf16)w1.z; pk.h[7] = (__bf16)w1.w;
      *reinterpret_cast<v8bf*>(&Bs[nn * 40 + kk0]) = pk.v;
    }
    __syncthreads();

    v16bf af[2];
#pragma unroll
    for (int mi = 0; mi < 2; ++mi)
      af[mi] = frag_a(&As[(mw + mi * 16) * 40], 40);
#pragma unroll
    for (int ni = 0; ni < 2; ++ni) {
      const v16bf bf = frag_b(&Bs[(nw + ni * 16) * 40], 40);
#pragma unroll
      for (int mi = 0; mi < 2; ++mi)
        acc[mi][ni] = wmma_bf16(af[mi], bf, acc[mi][ni]);
    }
    __syncthreads();
  }

#pragma unroll
  for (int mi = 0; mi < 2; ++mi)
#pragma unroll
    for (int ni = 0; ni < 2; ++ni) {
      const int col = nblk + nw + ni * 16 + (lane & 15);
      const float bcol = bo[col];
#pragma unroll
      for (int i = 0; i < 8; ++i) {
        const int row = mblk + mw + mi * 16 + (lane >> 4) * 8 + i;
        Out[(size_t)row * DD + col] = acc[mi][ni][i] + bcol;
      }
    }
}

// ---------------------------------------------------------------------------
// Host launcher.
// ---------------------------------------------------------------------------
extern "C" void kernel_launch(void* const* d_in, const int* in_sizes, int n_in,
                              void* d_out, int out_size, void* d_ws,
                              size_t ws_size, hipStream_t stream) {
  const float* x    = (const float*)d_in[0];
  const float* kv   = (const float*)d_in[1];
  const float* vv   = (const float*)d_in[2];
  const float* Wq   = (const float*)d_in[3];
  const float* bq   = (const float*)d_in[4];
  const float* Wk   = (const float*)d_in[5];
  const float* bk   = (const float*)d_in[6];
  const float* Wv   = (const float*)d_in[7];
  const float* bv   = (const float*)d_in[8];
  const float* Wo   = (const float*)d_in[9];
  const float* bo   = (const float*)d_in[10];
  const int*   mask = (const int*)d_in[11];
  float* out = (float*)d_out;

  const size_t nElem = (size_t)TT * BB * DD;  // 8,388,608 bf16 elems / buffer
  __bf16* Qws = (__bf16*)d_ws;
  __bf16* Kws = Qws + nElem;
  __bf16* Vws = Kws + nElem;
  __bf16* Aws = Vws + nElem;  // total 64 MB of workspace

  const dim3 gGemm(64, 16);
  proj_gemm_kernel<<<gGemm, 256, 0, stream>>>(x,  Wq, bq, Qws);
  proj_gemm_kernel<<<gGemm, 256, 0, stream>>>(kv, Wk, bk, Kws);
  proj_gemm_kernel<<<gGemm, 256, 0, stream>>>(vv, Wv, bv, Vws);
  attn_kernel<<<dim3(8, 128), 128, 0, stream>>>(Qws, Kws, Vws, Aws, mask);
  out_gemm_kernel<<<gGemm, 256, 0, stream>>>(Aws, Wo, bo, out);
}
